// GATv2_GCN_CNN_BiLSTM_Transformer_19009525252351
// MI455X (gfx1250) — matrix operand order repaired and checked by reading, not compiled, not run on hardware
//
#include <hip/hip_runtime.h>
#include <math.h>

#define NNODES 32768
#define NGRAPH 256
#define NPG    128
#define FDIM   78
#define NHEAD  10
#define D10    780
#define EBASE  131072
#define ETOT   163840
#define EDIM   128
#define SEQL   64
#define HID    64

typedef __attribute__((ext_vector_type(16))) _Float16 v16h;
typedef __attribute__((ext_vector_type(8)))  _Float16 v8h;
typedef __attribute__((ext_vector_type(8)))  float    v8f;

__device__ __forceinline__ float sigm(float x) { return 1.f / (1.f + expf(-x)); }

// ---------------------------------------------------------------------------
// WMMA GEMM, all-f16 operands:  C[M,N] = act( A[M,KP] @ B^T + bias )
// A: f16 [M, lda] (lda = K padded to mult of 32, zero-filled tail)
// B: f16 [N, ldb] (same padding) ; B(n,k) = Bp[n*ldb + k]
// One wave per 16(M) x 64(N) macro-tile, 4 v_wmma per 32-wide K chunk.
// Branch-free: fragments are aligned contiguous b128 loads (lda/ldb mult of
// 32 halves -> 64B rows), edges handled by pointer clamping (clamped lanes
// only feed never-stored outputs).  Optional f16 mirror output C16 feeds the
// next GEMM without an f32 round-trip.
// ---------------------------------------------------------------------------
__global__ void gemm_wmma_f16(const _Float16* __restrict__ A, int lda,
                              const _Float16* __restrict__ Bp, int ldb,
                              const float* __restrict__ bias,
                              float* __restrict__ C, int ldc,
                              _Float16* __restrict__ C16, int ldc16,
                              int M, int N, int KP, int act)
{
  const int lane = threadIdx.x;        // full wave32, EXEC all ones at wmma
  const int half = lane >> 4;
  const int l15  = lane & 15;
  const int tM   = blockIdx.y;
  const int tN4  = blockIdx.x;

  const int mA  = tM * 16 + l15;
  const int mAc = (mA < M) ? mA : (M - 1);
  const _Float16* pA = A + (size_t)mAc * lda + 8 * half;

  int nB[4];
  const _Float16* pB[4];
#pragma unroll
  for (int u = 0; u < 4; ++u) {
    nB[u] = tN4 * 64 + u * 16 + l15;
    int nc = (nB[u] < N) ? nB[u] : (N - 1);
    pB[u] = Bp + (size_t)nc * ldb + 8 * half;
  }

  v8f acc[4] = {v8f{}, v8f{}, v8f{}, v8f{}};

  for (int k0 = 0; k0 < KP; k0 += 32) {
    v8h a0 = *(const v8h*)(pA + k0);          // K run  k0+8h .. +7   (16B aligned)
    v8h a1 = *(const v8h*)(pA + k0 + 16);     // K run  k0+16+8h .. +7
    v16h a = __builtin_shufflevector(a0, a1, 0, 1, 2, 3, 4, 5, 6, 7,
                                     8, 9, 10, 11, 12, 13, 14, 15);
#pragma unroll
    for (int u = 0; u < 4; ++u) {
      v8h b0 = *(const v8h*)(pB[u] + k0);     // K run  k0+8h .. +7
      v8h b1 = *(const v8h*)(pB[u] + k0 + 16);
      v16h b = __builtin_shufflevector(b0, b1, 0, 1, 2, 3, 4, 5, 6, 7,
                                       8, 9, 10, 11, 12, 13, 14, 15);
      acc[u] = __builtin_amdgcn_wmma_f32_16x16x32_f16(false, a, false, b,
                                                      (short)0, acc[u], false, false);
    }
  }

#pragma unroll
  for (int u = 0; u < 4; ++u) {
    if (nB[u] < N) {
      float bv = bias ? bias[nB[u]] : 0.f;
#pragma unroll
      for (int i = 0; i < 8; ++i) {           // C/D: VGPR i -> row i + 8*half
        int m = tM * 16 + i + 8 * half;
        if (m < M) {
          float v = acc[u][i] + bv;
          if (act == 1) v = fmaxf(v, 0.f);
          if (C)   C[(size_t)m * ldc + nB[u]] = v;
          if (C16) C16[(size_t)m * ldc16 + nB[u]] = (_Float16)v;
        }
      }
    }
  }
}

// ---------------------------------------------------------------------------
// Staging / elementwise / graph kernels
// ---------------------------------------------------------------------------
__global__ void k_zero(float* p, size_t n) {
  size_t i = (size_t)blockIdx.x * blockDim.x + threadIdx.x;
  if (i < n) p[i] = 0.f;
}

// f32 [rows,ldsrc] -> f16 [rows,lddst], zero-padding cols K..lddst
__global__ void k_cvt16(const float* __restrict__ src, int ldsrc,
                        _Float16* __restrict__ dst, int lddst, int rows, int K)
{
  size_t i = (size_t)blockIdx.x * blockDim.x + threadIdx.x;
  if (i >= (size_t)rows * lddst) return;
  int r = (int)(i / lddst), k = (int)(i % lddst);
  dst[i] = (_Float16)((k < K) ? src[(size_t)r * ldsrc + k] : 0.f);
}

__global__ void k_deg(const int* __restrict__ ei, float* __restrict__ deg) {
  int e = blockIdx.x * blockDim.x + threadIdx.x;
  if (e >= ETOT) return;
  int d = (e < EBASE) ? ei[EBASE + e] : (e - EBASE);
  atomicAdd(deg + d, 1.f);
}

__global__ void k_dinv(float* deg, int n) {
  int i = blockIdx.x * blockDim.x + threadIdx.x;
  if (i < n) deg[i] = rsqrtf(fmaxf(deg[i], 1.f));
}

__global__ void k_gat_logit(const int* __restrict__ ei, const float* __restrict__ xl,
                            const float* __restrict__ xr, const float* __restrict__ att,
                            float* __restrict__ logit, unsigned* __restrict__ mEnc)
{
  int idx = blockIdx.x * blockDim.x + threadIdx.x;       // e*NHEAD + h
  if (idx >= ETOT * NHEAD) return;
  int e = idx / NHEAD, h = idx % NHEAD;
  int s, d;
  if (e < EBASE) { s = ei[e]; d = ei[EBASE + e]; } else { s = d = e - EBASE; }
  const float* pl = xl + (size_t)s * D10 + h * FDIM;
  const float* pr = xr + (size_t)d * D10 + h * FDIM;
  const float* pa = att + h * FDIM;
  float acc = 0.f;
  for (int c = 0; c < FDIM; ++c) {
    float v = pl[c] + pr[c];
    v = v > 0.f ? v : 0.2f * v;                           // leaky_relu(., 0.2)
    acc += pa[c] * v;
  }
  logit[idx] = acc;
  unsigned k = __float_as_uint(acc);
  k = (k & 0x80000000u) ? ~k : (k | 0x80000000u);         // order-preserving encode
  atomicMax(mEnc + (size_t)d * NHEAD + h, k);
}

__global__ void k_gat_w(const int* __restrict__ ei, float* __restrict__ logit,
                        const unsigned* __restrict__ mEnc, float* __restrict__ ssum)
{
  int idx = blockIdx.x * blockDim.x + threadIdx.x;
  if (idx >= ETOT * NHEAD) return;
  int e = idx / NHEAD, h = idx % NHEAD;
  int d = (e < EBASE) ? ei[EBASE + e] : (e - EBASE);
  unsigned enc = mEnc[(size_t)d * NHEAD + h];
  float m = (enc & 0x80000000u) ? __uint_as_float(enc & 0x7fffffffu) : __uint_as_float(~enc);
  float w = expf(logit[idx] - m);
  logit[idx] = w;
  atomicAdd(ssum + (size_t)d * NHEAD + h, w);
}

__global__ void k_gat_scatter(const int* __restrict__ ei, const float* __restrict__ w,
                              const float* __restrict__ ssum, const float* __restrict__ xl,
                              float* __restrict__ outp)
{
  size_t idx = (size_t)blockIdx.x * blockDim.x + threadIdx.x;  // e*D10 + cf
  if (idx >= (size_t)ETOT * D10) return;
  int e = (int)(idx / D10), cf = (int)(idx % D10);
  int h = cf / FDIM;
  int s, d;
  if (e < EBASE) { s = ei[e]; d = ei[EBASE + e]; } else { s = d = e - EBASE; }
  float alpha = w[(size_t)e * NHEAD + h] / ssum[(size_t)d * NHEAD + h];
  atomicAdd(outp + (size_t)d * D10 + cf, alpha * xl[(size_t)s * D10 + cf]);
}

// bias+relu in place (f32 only)
__global__ void k_bias_relu(float* __restrict__ xp, const float* __restrict__ b,
                            int C, size_t n)
{
  size_t i = (size_t)blockIdx.x * blockDim.x + threadIdx.x;
  if (i >= n) return;
  float v = xp[i] + b[i % C];
  xp[i] = v > 0.f ? v : 0.f;
}

// bias+relu in place + padded f16 mirror for the next GEMM's A operand
__global__ void k_bias_relu16(float* __restrict__ xp, const float* __restrict__ b, int C,
                              _Float16* __restrict__ x16, int ld16, int rows)
{
  size_t i = (size_t)blockIdx.x * blockDim.x + threadIdx.x;
  if (i >= (size_t)rows * ld16) return;
  int r = (int)(i / ld16), c = (int)(i % ld16);
  float v = 0.f;
  if (c < C) {
    v = xp[(size_t)r * C + c] + b[c];
    v = fmaxf(v, 0.f);
    xp[(size_t)r * C + c] = v;
  }
  x16[i] = (_Float16)v;
}

__global__ void k_gcn_scatter(const int* __restrict__ ei, const float* __restrict__ dinv,
                              const float* __restrict__ h, float* __restrict__ outp, int C)
{
  size_t idx = (size_t)blockIdx.x * blockDim.x + threadIdx.x;  // e*C + c
  if (idx >= (size_t)ETOT * C) return;
  int e = (int)(idx / C), c = (int)(idx % C);
  int s, d;
  if (e < EBASE) { s = ei[e]; d = ei[EBASE + e]; } else { s = d = e - EBASE; }
  atomicAdd(outp + (size_t)d * C + c, dinv[s] * dinv[d] * h[(size_t)s * C + c]);
}

// batch is block-contiguous (128 nodes/graph); writes straight into f16 xg
__global__ void k_pool16(const float* __restrict__ x, int C,
                         _Float16* __restrict__ xg, int colBase)
{
  int idx = blockIdx.x * blockDim.x + threadIdx.x;       // g*C + c
  if (idx >= NGRAPH * C) return;
  int g = idx / C, c = idx % C;
  const float* p = x + ((size_t)g * NPG) * C + c;
  float mx = -3.4e38f, sm = 0.f;
  for (int i = 0; i < NPG; ++i) { float v = p[(size_t)i * C]; mx = fmaxf(mx, v); sm += v; }
  _Float16* row = xg + (size_t)g * 2208 + colBase;
  row[c] = (_Float16)mx;
  row[C + c] = (_Float16)(sm / (float)NPG);
}

// Xmat16[(b*128+h), c] = emb[target[b,c], h], f16, K padded 1000->1024
__global__ void k_xmat16(const int* __restrict__ tgt, const float* __restrict__ E,
                         _Float16* __restrict__ Xm)
{
  int idx = blockIdx.x * blockDim.x + threadIdx.x;       // b*1024 + c
  if (idx >= NGRAPH * 1024) return;
  int b = idx / 1024, c = idx % 1024;
  if (c < 1000) {
    const float* e = E + (size_t)tgt[b * 1000 + c] * EDIM;
    _Float16* o = Xm + ((size_t)b * EDIM) * 1024 + c;
    for (int hh = 0; hh < EDIM; ++hh) o[(size_t)hh * 1024] = (_Float16)e[hh];
  } else {
    _Float16* o = Xm + ((size_t)b * EDIM) * 1024 + c;
    for (int hh = 0; hh < EDIM; ++hh) o[(size_t)hh * 1024] = (_Float16)0.f;
  }
}

// conv weight repack: W[o,c,kw] -> f16 Wt[kw][o][c], K padded 1000->1024
__global__ void k_convrepack16(const float* __restrict__ W, _Float16* __restrict__ Wt)
{
  int idx = blockIdx.x * blockDim.x + threadIdx.x;       // kw*131072 + o*1024 + c
  if (idx >= 3 * EDIM * 1024) return;
  int kw = idx / (EDIM * 1024);
  int r  = idx % (EDIM * 1024);
  int o  = r / 1024, c = r % 1024;
  Wt[idx] = (c < 1000) ? (_Float16)W[(size_t)o * 3000 + c * 3 + kw] : (_Float16)0.f;
}

// combine 3 shifted GEMM outputs + bias + relu + maxpool(2) + transpose -> f16 seq[S,B,128]
__global__ void k_convpool(const float* __restrict__ Y, const float* __restrict__ cb,
                           _Float16* __restrict__ seq)
{
  int idx = blockIdx.x * blockDim.x + threadIdx.x;       // b*128*64 + o*64 + j
  if (idx >= NGRAPH * EDIM * SEQL) return;
  int j = idx % SEQL;
  int o = (idx / SEQL) % EDIM;
  int b = idx / (SEQL * EDIM);
  float m = -3.4e38f;
  for (int q = 0; q < 2; ++q) {
    int hh = 2 * j + q;
    size_t base = ((size_t)b * EDIM + hh) * 384;
    float v = cb[o] + Y[base + 128 + o];                 // kw=1 (center)
    if (hh > 0)   v += Y[base - 384 + o];                // kw=0 at h-1
    if (hh < 127) v += Y[base + 384 + 256 + o];          // kw=2 at h+1
    v = v > 0.f ? v : 0.f;
    m = fmaxf(m, v);
  }
  seq[((size_t)j * NGRAPH + b) * EDIM + o] = (_Float16)m;
}

__global__ void k_addvec(const float* a, const float* b, float* c, int n) {
  int i = blockIdx.x * blockDim.x + threadIdx.x;
  if (i < n) c[i] = a[i] + b[i];
}

// fused LSTM pointwise update; writes f16 recurrent state + f32/f16 outputs
__global__ void k_lstm_step(const float* __restrict__ xw, const float* __restrict__ hw,
                            _Float16* __restrict__ h16, float* __restrict__ c,
                            float* __restrict__ yout, _Float16* __restrict__ yout16)
{
  int idx = blockIdx.x * blockDim.x + threadIdx.x;       // b*64 + j
  if (idx >= NGRAPH * HID) return;
  int b = idx / HID, j = idx % HID;
  const float* gx = xw + (size_t)b * 256;
  const float* gh = hw + (size_t)b * 256;
  float gi = gx[j]           + gh[j];
  float gf = gx[HID + j]     + gh[HID + j];
  float gg = gx[2 * HID + j] + gh[2 * HID + j];
  float go = gx[3 * HID + j] + gh[3 * HID + j];
  float cc = sigm(gf) * c[idx] + sigm(gi) * tanhf(gg);
  float hh = sigm(go) * tanhf(cc);
  c[idx] = cc;
  h16[idx] = (_Float16)hh;
  yout[(size_t)b * EDIM + j] = hh;                       // dir offset folded into ptr
  yout16[(size_t)b * EDIM + j] = (_Float16)hh;
}

__global__ void k_scores(const float* __restrict__ qkv, float* __restrict__ sc)
{
  int idx = blockIdx.x * blockDim.x + threadIdx.x;       // ((b*4+h)*64+s)*64 + t
  if (idx >= NGRAPH * 4 * SEQL * SEQL) return;
  int t = idx & 63, s = (idx >> 6) & 63, h = (idx >> 12) & 3, b = idx >> 14;
  const float* q = qkv + ((size_t)s * NGRAPH + b) * 384 + h * 32;
  const float* k = qkv + ((size_t)t * NGRAPH + b) * 384 + 128 + h * 32;
  float a = 0.f;
  for (int d = 0; d < 32; ++d) a += q[d] * k[d];
  sc[idx] = a * 0.17677669529663687f;                    // 1/sqrt(32)
}

__global__ void k_softmax64(float* __restrict__ sc)
{
  int row = blockIdx.x * blockDim.x + threadIdx.x;       // b*4*64 rows
  if (row >= NGRAPH * 4 * SEQL) return;
  float* p = sc + (size_t)row * SEQL;
  float m = -3.4e38f;
  for (int t = 0; t < SEQL; ++t) m = fmaxf(m, p[t]);
  float s = 0.f;
  for (int t = 0; t < SEQL; ++t) { float e = expf(p[t] - m); p[t] = e; s += e; }
  float inv = 1.f / s;
  for (int t = 0; t < SEQL; ++t) p[t] *= inv;
}

__global__ void k_attnout(const float* __restrict__ sc, const float* __restrict__ qkv,
                          _Float16* __restrict__ o16)
{
  int idx = blockIdx.x * blockDim.x + threadIdx.x;       // ((s*256)+b)*128 + h*32+d
  if (idx >= SEQL * NGRAPH * EDIM) return;
  int d = idx & 31, h = (idx >> 5) & 3, b = (idx >> 7) & 255, s = idx >> 15;
  const float* a = sc + (((size_t)b * 4 + h) * SEQL + s) * SEQL;
  float acc = 0.f;
  for (int t = 0; t < SEQL; ++t)
    acc += a[t] * qkv[((size_t)t * NGRAPH + b) * 384 + 256 + h * 32 + d];
  o16[idx] = (_Float16)acc;
}

__global__ void k_ln_res(float* __restrict__ y, const float* __restrict__ a,
                         const float* __restrict__ g, const float* __restrict__ b,
                         _Float16* __restrict__ y16)
{
  int r = blockIdx.x * blockDim.x + threadIdx.x;         // 16384 rows of 128
  if (r >= SEQL * NGRAPH) return;
  float* py = y + (size_t)r * EDIM;
  const float* pa = a + (size_t)r * EDIM;
  _Float16* p16 = y16 + (size_t)r * EDIM;
  float m = 0.f;
  for (int c = 0; c < EDIM; ++c) m += py[c] + pa[c];
  m *= (1.f / EDIM);
  float v = 0.f;
  for (int c = 0; c < EDIM; ++c) { float d = py[c] + pa[c] - m; v += d * d; }
  v *= (1.f / EDIM);
  float inv = rsqrtf(v + 1e-5f);
  for (int c = 0; c < EDIM; ++c) {
    float o = (py[c] + pa[c] - m) * inv * g[c] + b[c];
    py[c] = o;
    p16[c] = (_Float16)o;
  }
}

__global__ void k_meanS16(const float* __restrict__ y, _Float16* __restrict__ xc16)
{
  int idx = blockIdx.x * blockDim.x + threadIdx.x;       // b*128 + o
  if (idx >= NGRAPH * EDIM) return;
  float s = 0.f;
  for (int t = 0; t < SEQL; ++t) s += y[(size_t)t * NGRAPH * EDIM + idx];
  xc16[(size_t)(idx / EDIM) * 256 + 128 + (idx % EDIM)] = (_Float16)(s * (1.f / SEQL));
}

// ---------------------------------------------------------------------------
extern "C" void kernel_launch(void* const* d_in, const int* in_sizes, int n_in,
                              void* d_out, int out_size, void* d_ws, size_t ws_size,
                              hipStream_t stream)
{
  (void)out_size; (void)ws_size;
  if (n_in < 71) return;

  // Detect flattening: (A) insertion-order top-level + sorted params pytree,
  // (B) fully sorted pytree: batch, edge_index, params..., target, x.
  int xi, eii, ti, PB;
  if (in_sizes[0] == NNODES * FDIM) { xi = 0; eii = 1; ti = 3; PB = 4; }
  else                              { xi = 70; eii = 1; ti = 69; PB = 2; }

  const float* x   = (const float*)d_in[xi];
  const int*   ei  = (const int*)  d_in[eii];
  const int*   tgt = (const int*)  d_in[ti];
  auto P = [&](int i) { return (const float*)d_in[PB + i]; };

  const float *convW = P(0), *convB = P(1), *embT = P(2);
  const float *fc1W = P(3), *fc1B = P(4), *fc2W = P(5), *fc2B = P(6);
  const float *fg1W = P(7), *fg1B = P(8), *fg2W = P(9), *fg2B = P(10);
  const float *gWl = P(11), *gWr = P(12), *gAtt = P(13), *gBias = P(14), *gBl = P(15), *gBr = P(16);
  const float *gc1W = P(17), *gc1B = P(18), *gc2W = P(19), *gc2B = P(20);
  const float *gc3W = P(21), *gc3B = P(22), *gc4W = P(23), *gc4B = P(24);
  const float *outW = P(41), *outB = P(42);

  // ---- workspace layout ----
  float* ws = (float*)d_ws;
  size_t off = 0;
  auto alloc  = [&](size_t n) { size_t o = off; off += (n + 63) & ~(size_t)63; return ws + o; };
  auto allocH = [&](size_t nh) { return (_Float16*)alloc((nh + 1) / 2); };
  const size_t SLAB = (size_t)NNODES * D10;              // 25,559,040 floats
  float* S0 = alloc(4 * SLAB);                           // arena (graph slabs, reused later)
  float* S1 = S0 + SLAB;
  float* S2 = S1 + SLAB;
  float* S3 = S2 + SLAB;
  float* mEncF = alloc((size_t)NNODES * NHEAD);  unsigned* mEnc = (unsigned*)mEncF;
  float* ssum  = alloc((size_t)NNODES * NHEAD);
  float* wlog  = alloc((size_t)ETOT * NHEAD);
  float* deg   = alloc(NNODES);
  float* XW    = alloc((size_t)SEQL * NGRAPH * 256);
  float* HWb   = alloc((size_t)NGRAPH * 256);
  float* cst   = alloc((size_t)NGRAPH * HID);
  float* bsum  = alloc(256);
  float* qkv   = alloc((size_t)SEQL * NGRAPH * 384);
  float* scor  = alloc((size_t)NGRAPH * 4 * SEQL * SEQL);
  float* aprj  = alloc((size_t)SEQL * NGRAPH * EDIM);
  float* Y1    = alloc((size_t)SEQL * NGRAPH * EDIM);    // f32 LSTM outputs
  float* Y2    = alloc((size_t)SEQL * NGRAPH * EDIM);
  // f16 stages
  _Float16* x16    = allocH((size_t)NNODES * 96);
  _Float16* WL16   = allocH((size_t)D10 * 96);
  _Float16* WR16   = allocH((size_t)D10 * 96);
  _Float16* gc1W16 = allocH((size_t)D10 * 800);
  _Float16* gc2W16 = allocH((size_t)FDIM * 96);
  _Float16* gc3W16 = allocH((size_t)156 * 96);
  _Float16* gc4W16 = allocH((size_t)312 * 160);
  _Float16* fg1W16 = allocH((size_t)1500 * 2208);
  _Float16* fg2W16 = allocH((size_t)EDIM * 1504);
  _Float16* a78016 = allocH((size_t)NNODES * 800);       // GAT output (relu) f16
  _Float16* x2a16  = allocH((size_t)NNODES * 96);
  _Float16* x2b16  = allocH((size_t)NNODES * 160);
  _Float16* xg16   = allocH((size_t)NGRAPH * 2208);
  _Float16* g1o16  = allocH((size_t)NGRAPH * 1504);
  _Float16* xc16   = allocH((size_t)NGRAPH * 256);
  _Float16* f1o16  = allocH((size_t)NGRAPH * 1024);
  _Float16* f2o16  = allocH((size_t)NGRAPH * 256);
  _Float16* wrep16 = allocH((size_t)3 * EDIM * 1024);
  _Float16* seq16  = allocH((size_t)SEQL * NGRAPH * EDIM);
  _Float16* Y1_16  = allocH((size_t)SEQL * NGRAPH * EDIM);
  _Float16* Y2_16  = allocH((size_t)SEQL * NGRAPH * EDIM);
  _Float16* h16    = allocH((size_t)NGRAPH * HID);
  _Float16* atno16 = allocH((size_t)SEQL * NGRAPH * EDIM);
  _Float16* Wih16  = allocH((size_t)256 * 128);
  _Float16* Whh16  = allocH((size_t)256 * 64);
  _Float16* Wqkv16 = allocH((size_t)384 * 128);
  _Float16* Wo16   = allocH((size_t)128 * 128);
  _Float16* ff1W16 = allocH((size_t)2048 * 128);
  _Float16* ff2W16 = allocH((size_t)128 * 2048);
  _Float16* fc1W16 = allocH((size_t)1024 * 256);
  _Float16* fc2W16 = allocH((size_t)256 * 1024);
  _Float16* outW16 = allocH(256);
  // protein-phase aliases into the (then-dead) graph arena (16B aligned):
  _Float16* Xmat16 = (_Float16*)S0;                      // 32768*1024 halves
  _Float16* ffh16  = (_Float16*)(S0 + 17000000);         // 16384*2048 halves
  float*    Ybuf   = S0 + 35000000;                      // 32768*384 f32

  auto gemm = [&](const _Float16* A, int lda, const _Float16* B, int ldb,
                  const float* bias, float* C, int ldc, _Float16* C16, int ldc16,
                  int M, int N, int KP, int act) {
    dim3 g((unsigned)((N + 63) / 64), (unsigned)((M + 15) / 16));
    gemm_wmma_f16<<<g, dim3(32), 0, stream>>>(A, lda, B, ldb, bias, C, ldc,
                                              C16, ldc16, M, N, KP, act);
  };
  auto z = [&](float* p, size_t n) {
    k_zero<<<(unsigned)((n + 255) / 256), 256, 0, stream>>>(p, n);
  };
  auto cvt = [&](const float* src, int lds, _Float16* dst, int ldd, int rows, int K) {
    size_t n = (size_t)rows * ldd;
    k_cvt16<<<(unsigned)((n + 255) / 256), 256, 0, stream>>>(src, lds, dst, ldd, rows, K);
  };
  auto g1 = [&](size_t n) { return dim3((unsigned)((n + 255) / 256)); };

  // ================= stage weights to f16 (padded K) =================
  cvt(gWl, FDIM, WL16, 96, D10, FDIM);
  cvt(gWr, FDIM, WR16, 96, D10, FDIM);
  cvt(gc1W, D10, gc1W16, 800, D10, D10);
  cvt(gc2W, FDIM, gc2W16, 96, FDIM, FDIM);
  cvt(gc3W, FDIM, gc3W16, 96, 156, FDIM);
  cvt(gc4W, 156, gc4W16, 160, 312, 156);
  cvt(fg1W, 2184, fg1W16, 2208, 1500, 2184);
  cvt(fg2W, 1500, fg2W16, 1504, EDIM, 1500);
  cvt(fc1W, 256, fc1W16, 256, 1024, 256);
  cvt(fc2W, 1024, fc2W16, 1024, 256, 1024);
  cvt(outW, 256, outW16, 256, 1, 256);
  cvt(x, FDIM, x16, 96, NNODES, FDIM);

  // ================= graph branch =================
  z(mEncF, (size_t)NNODES * NHEAD);
  z(ssum,  (size_t)NNODES * NHEAD);
  z(deg, NNODES);
  z(S2, (size_t)NNODES * D10);                           // GAT accumulator
  gemm(x16, 96, WL16, 96, gBl, S0, D10, nullptr, 0, NNODES, D10, 96, 0);  // xl
  gemm(x16, 96, WR16, 96, gBr, S1, D10, nullptr, 0, NNODES, D10, 96, 0);  // xr
  k_deg<<<g1(ETOT), 256, 0, stream>>>(ei, deg);
  k_dinv<<<g1(NNODES), 256, 0, stream>>>(deg, NNODES);
  k_gat_logit<<<g1((size_t)ETOT * NHEAD), 256, 0, stream>>>(ei, S0, S1, gAtt, wlog, mEnc);
  k_gat_w<<<g1((size_t)ETOT * NHEAD), 256, 0, stream>>>(ei, wlog, mEnc, ssum);
  k_gat_scatter<<<g1((size_t)ETOT * D10), 256, 0, stream>>>(ei, wlog, ssum, S0, S2);
  k_bias_relu16<<<g1((size_t)NNODES * 800), 256, 0, stream>>>(S2, gBias, D10, a78016, 800, NNODES);
  // GCN1 (780 -> 780)
  gemm(a78016, 800, gc1W16, 800, nullptr, S3, D10, nullptr, 0, NNODES, D10, 800, 0);
  z(S1, (size_t)NNODES * D10);
  k_gcn_scatter<<<g1((size_t)ETOT * D10), 256, 0, stream>>>(ei, deg, S3, S1, D10);
  k_bias_relu<<<g1((size_t)NNODES * D10), 256, 0, stream>>>(S1, gc1B, D10, (size_t)NNODES * D10);
  z((float*)xg16, (size_t)NGRAPH * 2208 / 2);            // zero-pad xg16 tail cols
  k_pool16<<<g1((size_t)NGRAPH * D10), 256, 0, stream>>>(S1, D10, xg16, 0);
  // GCN2/3/4 chain on original x
  gemm(x16, 96, gc2W16, 96, nullptr, S0, FDIM, nullptr, 0, NNODES, FDIM, 96, 0);
  z(S2, (size_t)NNODES * FDIM);
  k_gcn_scatter<<<g1((size_t)ETOT * FDIM), 256, 0, stream>>>(ei, deg, S0, S2, FDIM);
  k_bias_relu16<<<g1((size_t)NNODES * 96), 256, 0, stream>>>(S2, gc2B, FDIM, x2a16, 96, NNODES);
  gemm(x2a16, 96, gc3W16, 96, nullptr, S0, 156, nullptr, 0, NNODES, 156, 96, 0);
  z(S3, (size_t)NNODES * 156);
  k_gcn_scatter<<<g1((size_t)ETOT * 156), 256, 0, stream>>>(ei, deg, S0, S3, 156);
  k_bias_relu16<<<g1((size_t)NNODES * 160), 256, 0, stream>>>(S3, gc3B, 156, x2b16, 160, NNODES);
  gemm(x2b16, 160, gc4W16, 160, nullptr, S0, 312, nullptr, 0, NNODES, 312, 160, 0);
  z(S2, (size_t)NNODES * 312);
  k_gcn_scatter<<<g1((size_t)ETOT * 312), 256, 0, stream>>>(ei, deg, S0, S2, 312);
  k_bias_relu<<<g1((size_t)NNODES * 312), 256, 0, stream>>>(S2, gc4B, 312, (size_t)NNODES * 312);
  k_pool16<<<g1((size_t)NGRAPH * 312), 256, 0, stream>>>(S2, 312, xg16, 1560);
  // graph head
  z((float*)g1o16, (size_t)NGRAPH * 1504 / 2);           // zero-pad g1o16 tail cols
  gemm(xg16, 2208, fg1W16, 2208, fg1B, nullptr, 0, g1o16, 1504, NGRAPH, 1500, 2208, 1);
  gemm(g1o16, 1504, fg2W16, 1504, fg2B, nullptr, 0, xc16, 256, NGRAPH, 128, 1504, 0);

  // ================= protein branch =================
  k_xmat16<<<g1((size_t)NGRAPH * 1024), 256, 0, stream>>>(tgt, embT, Xmat16);
  k_convrepack16<<<g1((size_t)3 * EDIM * 1024), 256, 0, stream>>>(convW, wrep16);
  for (int kw = 0; kw < 3; ++kw)   // Conv1d as 3 shifted GEMMs
    gemm(Xmat16, 1024, wrep16 + (size_t)kw * EDIM * 1024, 1024, nullptr,
         Ybuf + kw * 128, 384, nullptr, 0, NNODES, 128, 1024, 0);
  k_convpool<<<g1((size_t)NGRAPH * EDIM * SEQL), 256, 0, stream>>>(Ybuf, convB, seq16);

  // 2-layer BiLSTM; sorted param order per layer: b-cell{Whh,Wih,bhh,bih}, f-cell{...}
  _Float16* Yin16 = seq16;
  float*    Yo32[2]  = {Y1, Y2};
  _Float16* Yo16[2]  = {Y1_16, Y2_16};
  for (int l = 0; l < 2; ++l) {
    for (int dir = 0; dir < 2; ++dir) {                  // dir 0 = forward('f'), 1 = backward('b')
      int cb = 25 + l * 8 + (dir == 0 ? 4 : 0);
      const float *Whh = P(cb + 0), *Wih = P(cb + 1), *bhh = P(cb + 2), *bih = P(cb + 3);
      cvt(Whh, HID, Whh16, HID, 256, HID);
      cvt(Wih, EDIM, Wih16, EDIM, 256, EDIM);
      k_addvec<<<1, 256, 0, stream>>>(bih, bhh, bsum, 256);
      gemm(Yin16, EDIM, Wih16, EDIM, bsum, XW, 256, nullptr, 0,
           SEQL * NGRAPH, 256, EDIM, 0);
      z((float*)h16, (size_t)NGRAPH * HID / 2);
      z(cst, (size_t)NGRAPH * HID);
      for (int stp = 0; stp < SEQL; ++stp) {
        int t = (dir == 0) ? stp : (SEQL - 1 - stp);
        gemm(h16, HID, Whh16, HID, nullptr, HWb, 256, nullptr, 0, NGRAPH, 256, HID, 0);
        k_lstm_step<<<g1((size_t)NGRAPH * HID), 256, 0, stream>>>(
            XW + (size_t)t * NGRAPH * 256, HWb, h16, cst,
            Yo32[l] + (size_t)t * NGRAPH * EDIM + dir * HID,
            Yo16[l] + (size_t)t * NGRAPH * EDIM + dir * HID);
      }
    }
    Yin16 = Yo16[l];
  }

  // 2 post-norm transformer layers on Y2 (param order: Wo,Wqkv,bo,bqkv,ff1,ff2,ln1b,ln1g,ln2b,ln2g)
  for (int l = 0; l < 2; ++l) {
    int tb = 43 + l * 12;
    const float *Wo = P(tb + 0), *Wqkv = P(tb + 1), *bo = P(tb + 2), *bqkv = P(tb + 3);
    const float *ff1Wp = P(tb + 4), *ff1b = P(tb + 5), *ff2Wp = P(tb + 6), *ff2b = P(tb + 7);
    const float *ln1b = P(tb + 8), *ln1g = P(tb + 9), *ln2b = P(tb + 10), *ln2g = P(tb + 11);
    cvt(Wqkv, EDIM, Wqkv16, EDIM, 384, EDIM);
    cvt(Wo, EDIM, Wo16, EDIM, EDIM, EDIM);
    cvt(ff1Wp, EDIM, ff1W16, EDIM, 2048, EDIM);
    cvt(ff2Wp, 2048, ff2W16, 2048, EDIM, 2048);
    gemm(Y2_16, EDIM, Wqkv16, EDIM, bqkv, qkv, 384, nullptr, 0,
         SEQL * NGRAPH, 384, EDIM, 0);
    k_scores<<<g1((size_t)NGRAPH * 4 * SEQL * SEQL), 256, 0, stream>>>(qkv, scor);
    k_softmax64<<<g1((size_t)NGRAPH * 4 * SEQL), 256, 0, stream>>>(scor);
    k_attnout<<<g1((size_t)SEQL * NGRAPH * EDIM), 256, 0, stream>>>(scor, qkv, atno16);
    gemm(atno16, EDIM, Wo16, EDIM, bo, aprj, EDIM, nullptr, 0,
         SEQL * NGRAPH, EDIM, EDIM, 0);
    k_ln_res<<<g1((size_t)SEQL * NGRAPH), 256, 0, stream>>>(Y2, aprj, ln1g, ln1b, Y2_16);
    gemm(Y2_16, EDIM, ff1W16, EDIM, ff1b, nullptr, 0, ffh16, 2048,
         SEQL * NGRAPH, 2048, EDIM, 1);
    gemm(ffh16, 2048, ff2W16, 2048, ff2b, aprj, EDIM, nullptr, 0,
         SEQL * NGRAPH, EDIM, 2048, 0);
    k_ln_res<<<g1((size_t)SEQL * NGRAPH), 256, 0, stream>>>(Y2, aprj, ln2g, ln2b, Y2_16);
  }
  k_meanS16<<<g1((size_t)NGRAPH * EDIM), 256, 0, stream>>>(Y2, xc16);  // -> xc16[:, 128:]

  // ================= head =================
  gemm(xc16, 256, fc1W16, 256, fc1B, nullptr, 0, f1o16, 1024, NGRAPH, 1024, 256, 1);
  gemm(f1o16, 1024, fc2W16, 1024, fc2B, nullptr, 0, f2o16, 256, NGRAPH, 256, 1024, 1);
  gemm(f2o16, 256, outW16, 256, outB, (float*)d_out, 1, nullptr, 0, NGRAPH, 1, 256, 0);
}
